// AutoregressiveAttentionalLSTM_43087111913938
// MI455X (gfx1250) — matrix operand order repaired and checked by reading, not compiled, not run on hardware
//
#include <hip/hip_runtime.h>
#include <hip/hip_bf16.h>

// Problem constants (match reference)
#define B_   16
#define S_   512
#define T_   128
#define E_   256
#define ENC_ 32
#define DEC_ 128
#define VOUT_ 32000

typedef __attribute__((ext_vector_type(16))) _Float16 v16h;
typedef __attribute__((ext_vector_type(8)))  _Float16 v8h;
typedef __attribute__((ext_vector_type(8)))  float    v8f;

union V16H { v16h v; v8h h[2]; };

__device__ __forceinline__ float sigm_(float x){ return 1.0f/(1.0f + __expf(-x)); }

// ---- WMMA operand loaders -------------------------------------------------
// A: row-major f16 (M x K). 16x32 tile, contiguous K order (matches Bpk):
//  lane group 0 (lanes 0-15): K = 0..15 ; group 1: K = 16..31
__device__ __forceinline__ v16h load_a16(const _Float16* __restrict__ A, int K,
                                         int mbase, int kt, int lane) {
  int m  = mbase + (lane & 15);
  int k0 = kt*32 + ((lane >> 4) << 4);
  V16H r;
  r.h[0] = *(const v8h*)(A + (size_t)m*K + k0);
  r.h[1] = *(const v8h*)(A + (size_t)m*K + k0 + 8);
  return r.v;
}
// B: pre-packed tiles; lane holds col n=lane&15, K-range (lane>>4)*16 + h,
// 16 halves contiguous per lane.
__device__ __forceinline__ v16h load_bpk(const _Float16* __restrict__ Bpk, int KT,
                                         int nt, int kt, int lane) {
  const _Float16* p = Bpk + ((((size_t)nt*KT + kt)*32 + lane) << 4);
  V16H r;
  r.h[0] = *(const v8h*)(p);
  r.h[1] = *(const v8h*)(p + 8);
  return r.v;
}

// ---- Weight pack: f32 row-major (K x N) -> f16 B-tile layout ---------------
__global__ __launch_bounds__(256) void pack_b_kernel(const float* __restrict__ W,
    _Float16* __restrict__ dst, int K, int N) {
  int idx = (int)(blockIdx.x * blockDim.x + threadIdx.x);
  if (idx >= K * N) return;
  int h  = idx & 15;
  int L  = (idx >> 4) & 31;
  int t  = idx >> 9;
  int KT = K >> 5;
  int kt = t % KT;
  int nt = t / KT;
  int k  = kt*32 + ((L >> 4) << 4) + h;
  int n  = (nt << 4) + (L & 15);
  dst[idx] = (_Float16)W[(size_t)k * N + n];
}

// ---- Embedding + sinusoidal positional encoding, store f16 -----------------
__global__ __launch_bounds__(256) void embed_src_kernel(const int* __restrict__ src,
    const float* __restrict__ emb, _Float16* __restrict__ dst) {
  int idx = (int)(blockIdx.x * blockDim.x + threadIdx.x);   // B*S*E
  int e  = idx & (E_-1);
  int bs = idx >> 8;
  int s  = bs & (S_-1);
  const float LR = 0.07195578f;  // ln(10000)/128
  float pe;
  if (e < 128) { float rate = __expf(-(float)e * LR);        pe = __sinf((float)s * rate); }
  else         { float rate = __expf(-(float)(e-128) * LR);  pe = __cosf((float)s * rate); }
  float v = emb[(size_t)src[bs]*E_ + e] * 16.0f + pe;        // sqrt(E)=16
  dst[idx] = (_Float16)v;
}

__global__ __launch_bounds__(256) void embed_tgt_kernel(const int* __restrict__ tgt,
    const float* __restrict__ emb, _Float16* __restrict__ dst) {
  int idx = (int)(blockIdx.x * blockDim.x + threadIdx.x);   // B*T*E
  int e  = idx & (E_-1);
  int bt = idx >> 8;
  dst[idx] = (_Float16)emb[(size_t)tgt[bt]*E_ + e];
}

// ---- WMMA GEMM: C(MxN) = A(MxK,f16) * Bpk + bias[n] ------------------------
template<bool NT>
__global__ __launch_bounds__(256) void gemm_bias_kernel(const _Float16* __restrict__ A,
    const _Float16* __restrict__ Bpk, const float* __restrict__ bias,
    float* __restrict__ C, int M, int N, int K, int nG) {
  int wave = (int)((blockIdx.x * blockDim.x + threadIdx.x) >> 5);
  int lane = threadIdx.x & 31;
  int mt = wave / nG;
  int ng = wave - mt * nG;
  if (mt >= (M >> 4)) return;              // wave-uniform (grids sized exactly)
  int KT = K >> 5;
  int mbase = mt << 4;
  int nt0 = ng << 2;
  v8f acc[4];
  #pragma unroll
  for (int j = 0; j < 4; ++j) {
    float bv = bias[(nt0 + j)*16 + (lane & 15)];
    #pragma unroll
    for (int r = 0; r < 8; ++r) acc[j][r] = bv;
  }
  for (int kt = 0; kt < KT; ++kt) {
    v16h a = load_a16(A, K, mbase, kt, lane);
    #pragma unroll
    for (int j = 0; j < 4; ++j) {
      v16h b = load_bpk(Bpk, KT, nt0 + j, kt, lane);
      acc[j] = __builtin_amdgcn_wmma_f32_16x16x32_f16(false, a, false, b,
                                                      (short)0, acc[j], false, false);
    }
  }
  int mrow = mbase + ((lane >> 4) << 3);
  #pragma unroll
  for (int j = 0; j < 4; ++j) {
    int n = (nt0 + j)*16 + (lane & 15);
    #pragma unroll
    for (int r = 0; r < 8; ++r) {
      float* p = C + (size_t)(mrow + r)*N + n;
      if (NT) __builtin_nontemporal_store(acc[j][r], p);
      else    *p = acc[j][r];
    }
  }
}

// ---- Decoder GEMM: z = temb@Wd2 + cz[b]  (cz folds ctx@Wd[:64] + bd) -------
__global__ __launch_bounds__(256) void gemm_dec_kernel(const _Float16* __restrict__ A,
    const _Float16* __restrict__ Bpk, const float* __restrict__ cz,
    float* __restrict__ C, int M, int N, int K, int nG) {
  int wave = (int)((blockIdx.x * blockDim.x + threadIdx.x) >> 5);
  int lane = threadIdx.x & 31;
  int mt = wave / nG;
  int ng = wave - mt * nG;
  if (mt >= (M >> 4)) return;
  int KT = K >> 5;
  int mbase = mt << 4;
  int nt0 = ng << 2;
  int b = mt >> 3;                          // T/16 = 8 row-tiles per batch
  v8f acc[4];
  #pragma unroll
  for (int j = 0; j < 4; ++j) {
    float bv = cz[(size_t)b*N + (nt0 + j)*16 + (lane & 15)];
    #pragma unroll
    for (int r = 0; r < 8; ++r) acc[j][r] = bv;
  }
  for (int kt = 0; kt < KT; ++kt) {
    v16h a = load_a16(A, K, mbase, kt, lane);
    #pragma unroll
    for (int j = 0; j < 4; ++j) {
      v16h bm = load_bpk(Bpk, KT, nt0 + j, kt, lane);
      acc[j] = __builtin_amdgcn_wmma_f32_16x16x32_f16(false, a, false, bm,
                                                      (short)0, acc[j], false, false);
    }
  }
  int mrow = mbase + ((lane >> 4) << 3);
  #pragma unroll
  for (int j = 0; j < 4; ++j) {
    int n = (nt0 + j)*16 + (lane & 15);
    #pragma unroll
    for (int r = 0; r < 8; ++r)
      C[(size_t)(mrow + r)*N + n] = acc[j][r];
  }
}

// ---- Sequential LSTM recurrence: one block per (dir, batch) ----------------
__global__ __launch_bounds__(128) void lstm_kernel(const float* __restrict__ XW,
    const float* __restrict__ Uf, const float* __restrict__ Ub,
    float* __restrict__ enc, float* __restrict__ hidden) {
  __shared__ float U[ENC_ * 4 * ENC_];   // 32x128
  __shared__ float hsh[ENC_];
  __shared__ float zsh[4 * ENC_];
  int dir = (int)(blockIdx.x >> 4);
  int b   = (int)(blockIdx.x & 15);
  int j   = threadIdx.x;                 // 0..127
  const float* Um = dir ? Ub : Uf;
  for (int i = j; i < ENC_ * 4 * ENC_; i += 128) U[i] = Um[i];
  if (j < ENC_) hsh[j] = 0.0f;
  __syncthreads();
  const float* xw = XW + ((size_t)dir * B_ * S_ + (size_t)b * S_) * (4 * ENC_);
  float c = 0.0f;
  for (int step = 0; step < S_; ++step) {
    int t = dir ? (S_ - 1 - step) : step;
    float z = xw[(size_t)t * (4 * ENC_) + j];
    #pragma unroll
    for (int k = 0; k < ENC_; ++k) z += hsh[k] * U[k * (4 * ENC_) + j];
    zsh[j] = z;
    __syncthreads();
    if (j < ENC_) {
      float gi = zsh[j], gf = zsh[ENC_ + j], gg = zsh[2*ENC_ + j], go = zsh[3*ENC_ + j];
      c = sigm_(gf) * c + sigm_(gi) * tanhf(gg);
      float h = sigm_(go) * tanhf(c);
      hsh[j] = h;
      enc[((size_t)b * S_ + t) * (2 * ENC_) + dir * ENC_ + j] = h;
    }
    __syncthreads();
  }
  if (j < ENC_) hidden[b * (2 * ENC_) + dir * ENC_ + j] = hsh[j];
}

// ---- q[b] = hidden[b]@W1 + b1 + b2 ----------------------------------------
__global__ __launch_bounds__(128) void qvec_kernel(const float* __restrict__ hidden,
    const float* __restrict__ W1, const float* __restrict__ b1,
    const float* __restrict__ b2, float* __restrict__ q) {
  int b = blockIdx.x, j = threadIdx.x;   // 128 cols
  float acc = b1[j] + b2[j];
  #pragma unroll 8
  for (int k = 0; k < 2*ENC_; ++k) acc += hidden[b * (2*ENC_) + k] * W1[k * DEC_ + j];
  q[b * DEC_ + j] = acc;
}

// ---- score[b,s] = Vw . tanh(q[b] + enc[b,s]@W2) + Vb ; one wave per s ------
__global__ __launch_bounds__(256) void score_kernel(const float* __restrict__ enc,
    const float* __restrict__ q, const float* __restrict__ W2,
    const float* __restrict__ Vw, const float* __restrict__ Vb,
    float* __restrict__ scores) {
  int b = (int)(blockIdx.x >> 6);
  int w = threadIdx.x >> 5;
  int s = ((int)(blockIdx.x & 63) << 3) + w;
  int lane = threadIdx.x & 31;
  __shared__ float erow[8][2*ENC_];
  const float* er = enc + ((size_t)b * S_ + s) * (2*ENC_);
  erow[w][lane]      = er[lane];
  erow[w][32 + lane] = er[32 + lane];
  __syncthreads();
  float part = 0.0f;
  #pragma unroll
  for (int cc = 0; cc < 4; ++cc) {
    int cidx = lane + cc * 32;
    float v = q[b * DEC_ + cidx];
    #pragma unroll 8
    for (int k = 0; k < 2*ENC_; ++k) v += erow[w][k] * W2[k * DEC_ + cidx];
    part += Vw[cidx] * tanhf(v);
  }
  #pragma unroll
  for (int off = 16; off; off >>= 1) part += __shfl_down(part, off, 32);
  if (lane == 0) scores[(size_t)b * S_ + s] = part + Vb[0];
}

// ---- softmax over S + context vector ---------------------------------------
__global__ __launch_bounds__(512) void softmax_ctx_kernel(const float* __restrict__ scores,
    const float* __restrict__ enc, float* __restrict__ ctx) {
  __shared__ float w[S_];
  __shared__ float red[S_];
  int b = blockIdx.x, s = threadIdx.x;
  float sc = scores[(size_t)b * S_ + s];
  red[s] = sc; __syncthreads();
  for (int off = 256; off; off >>= 1) { if (s < off) red[s] = fmaxf(red[s], red[s+off]); __syncthreads(); }
  float m = red[0]; __syncthreads();
  float e = __expf(sc - m);
  w[s] = e; red[s] = e; __syncthreads();
  for (int off = 256; off; off >>= 1) { if (s < off) red[s] += red[s+off]; __syncthreads(); }
  float inv = 1.0f / red[0];
  if (s < 2*ENC_) {
    float acc = 0.0f;
    for (int t = 0; t < S_; ++t) acc += w[t] * enc[((size_t)b * S_ + t) * (2*ENC_) + s];
    ctx[b * (2*ENC_) + s] = acc * inv;
  }
}

// ---- cz[b,n] = bd[n] + ctx[b]@Wd[0:64, n] ----------------------------------
__global__ __launch_bounds__(512) void cz_kernel(const float* __restrict__ ctx,
    const float* __restrict__ Wd, const float* __restrict__ bd, float* __restrict__ cz) {
  int b = blockIdx.x, n = threadIdx.x;   // 512 cols
  float acc = bd[n];
  #pragma unroll 8
  for (int k = 0; k < 2*ENC_; ++k) acc += ctx[b * (2*ENC_) + k] * Wd[(size_t)k * (4*DEC_) + n];
  cz[(size_t)b * (4*DEC_) + n] = acc;
}

// ---- decoder gates: h = sig(o)*tanh(sig(i)*tanh(g)), store f16 -------------
__global__ __launch_bounds__(256) void gates_kernel(const float* __restrict__ z,
    _Float16* __restrict__ h16) {
  int idx = (int)(blockIdx.x * blockDim.x + threadIdx.x);  // 2048*128
  int row = idx >> 7, j = idx & 127;
  const float* zr = z + (size_t)row * (4*DEC_);
  float gi = zr[j], gg = zr[2*DEC_ + j], go = zr[3*DEC_ + j];
  float c = sigm_(gi) * tanhf(gg);           // c_prev = 0: forget term vanishes
  h16[idx] = (_Float16)(sigm_(go) * tanhf(c));
}

// ---------------------------------------------------------------------------
extern "C" void kernel_launch(void* const* d_in, const int* in_sizes, int n_in,
                              void* d_out, int out_size, void* d_ws, size_t ws_size,
                              hipStream_t stream) {
  (void)in_sizes; (void)n_in; (void)out_size; (void)ws_size;
  const int*   source  = (const int*)  d_in[0];
  const int*   target  = (const int*)  d_in[1];
  const float* src_emb = (const float*)d_in[2];
  const float* tgt_emb = (const float*)d_in[3];
  const float* Wf  = (const float*)d_in[4];
  const float* Uf  = (const float*)d_in[5];
  const float* bf  = (const float*)d_in[6];
  const float* Wb  = (const float*)d_in[7];
  const float* Ub  = (const float*)d_in[8];
  const float* bb  = (const float*)d_in[9];
  const float* W1  = (const float*)d_in[10];
  const float* b1  = (const float*)d_in[11];
  const float* W2  = (const float*)d_in[12];
  const float* b2  = (const float*)d_in[13];
  const float* Vw  = (const float*)d_in[14];
  const float* Vb  = (const float*)d_in[15];
  const float* Wd  = (const float*)d_in[16];
  const float* bd  = (const float*)d_in[17];
  const float* Wfc = (const float*)d_in[18];
  const float* bfc = (const float*)d_in[19];
  float* out = (float*)d_out;

  // workspace carve-out (256B aligned)
  char* ws = (char*)d_ws;
  size_t off = 0;
  auto take = [&](size_t bytes) { char* p = ws + off; off += (bytes + 255) & ~(size_t)255; return p; };
  _Float16* src16   = (_Float16*)take((size_t)B_*S_*E_*2);        // 4 MB
  _Float16* Wf_pk   = (_Float16*)take((size_t)E_*4*ENC_*2);       // 64 KB
  _Float16* Wb_pk   = (_Float16*)take((size_t)E_*4*ENC_*2);
  float*    XW      = (float*)   take((size_t)2*B_*S_*4*ENC_*4);  // 8 MB (fwd|bwd)
  float*    enc     = (float*)   take((size_t)B_*S_*2*ENC_*4);    // 2 MB
  float*    hidden  = (float*)   take((size_t)B_*2*ENC_*4);
  float*    q       = (float*)   take((size_t)B_*DEC_*4);
  float*    scores  = (float*)   take((size_t)B_*S_*4);
  float*    ctx     = (float*)   take((size_t)B_*2*ENC_*4);
  float*    czb     = (float*)   take((size_t)B_*4*DEC_*4);
  _Float16* temb16  = (_Float16*)take((size_t)B_*T_*E_*2);        // 1 MB
  _Float16* Wd2_pk  = (_Float16*)take((size_t)E_*4*DEC_*2);       // 256 KB
  float*    zbuf    = (float*)   take((size_t)B_*T_*4*DEC_*4);    // 4 MB
  _Float16* h16     = (_Float16*)take((size_t)B_*T_*DEC_*2);      // 512 KB
  _Float16* Wfc_pk  = (_Float16*)take((size_t)DEC_*VOUT_*2);      // 8 MB

  // 1) weight packs (f32 -> f16 B-tile layout)
  pack_b_kernel<<<(E_*4*ENC_)/256, 256, 0, stream>>>(Wf, Wf_pk, E_, 4*ENC_);
  pack_b_kernel<<<(E_*4*ENC_)/256, 256, 0, stream>>>(Wb, Wb_pk, E_, 4*ENC_);
  pack_b_kernel<<<(E_*4*DEC_)/256, 256, 0, stream>>>(Wd + (size_t)(2*ENC_)*(4*DEC_), Wd2_pk, E_, 4*DEC_);
  pack_b_kernel<<<(DEC_*VOUT_)/256, 256, 0, stream>>>(Wfc, Wfc_pk, DEC_, VOUT_);

  // 2) source embedding + positional encoding (f16)
  embed_src_kernel<<<(B_*S_*E_)/256, 256, 0, stream>>>(source, src_emb, src16);

  // 3) parallel LSTM input projections: XW = src@W + b  (WMMA)
  //    M=8192, N=128, K=256 -> nG=2 -> 1024 waves -> 128 blocks
  gemm_bias_kernel<false><<<128, 256, 0, stream>>>(src16, Wf_pk, bf, XW,
                                                   B_*S_, 4*ENC_, E_, 2);
  gemm_bias_kernel<false><<<128, 256, 0, stream>>>(src16, Wb_pk, bb,
                                                   XW + (size_t)B_*S_*4*ENC_,
                                                   B_*S_, 4*ENC_, E_, 2);

  // 4) sequential recurrence (32 blocks: 2 dirs x 16 batches)
  lstm_kernel<<<32, 128, 0, stream>>>(XW, Uf, Ub, enc, hidden);

  // 5) attention
  qvec_kernel<<<B_, 128, 0, stream>>>(hidden, W1, b1, b2, q);
  score_kernel<<<B_*64, 256, 0, stream>>>(enc, q, W2, Vw, Vb, scores);
  softmax_ctx_kernel<<<B_, 512, 0, stream>>>(scores, enc, ctx);

  // 6) decoder: cz, target embedding, z GEMM, gates
  cz_kernel<<<B_, 512, 0, stream>>>(ctx, Wd, bd, czb);
  embed_tgt_kernel<<<(B_*T_*E_)/256, 256, 0, stream>>>(target, tgt_emb, temb16);
  //    M=2048, N=512, K=256 -> nG=8 -> 1024 waves -> 128 blocks
  gemm_dec_kernel<<<128, 256, 0, stream>>>(temb16, Wd2_pk, czb, zbuf,
                                           B_*T_, 4*DEC_, E_, 8);
  gates_kernel<<<(B_*T_*DEC_)/256, 256, 0, stream>>>(zbuf, h16);

  // 7) output projection (store-bound, non-temporal): M=2048, N=32000, K=128
  //    nG=500 -> 64000 waves -> 8000 blocks
  gemm_bias_kernel<true><<<8000, 256, 0, stream>>>(h16, Wfc_pk, bfc, out,
                                                   B_*T_, VOUT_, DEC_, 500);
}
